// Attention_53188874993896
// MI455X (gfx1250) — matrix operand order repaired and checked by reading, not compiled, not run on hardware
//
#include <hip/hip_runtime.h>
#include <hip/hip_bf16.h>

// Problem constants (fixed by the reference)
#define BB 64
#define SS 256
#define DD 256
#define HH 8
#define HD 32
#define P8 32
#define PD 8
#define MM (BB * SS)   // 16384 rows
#define BSTR 264       // padded LDS col stride in halfs (528B = 132 dwords, 132%64=4 -> conflict-free)

typedef _Float16 h16;
typedef __attribute__((ext_vector_type(8)))  _Float16 v8h;
typedef __attribute__((ext_vector_type(16))) _Float16 v16h;
typedef __attribute__((ext_vector_type(8)))  float    v8f;

static __device__ inline v16h cat8(v8h lo, v8h hi) {
  return __builtin_shufflevector(lo, hi, 0,1,2,3,4,5,6,7,8,9,10,11,12,13,14,15);
}

// A fragment (16x32 f16, MxK): lane L<16 -> row L, K = {0..7, 16..23};
// lane L>=16 -> row L-16, K = {8..15, 24..31}.   p = &A[row*lda + k0]
static __device__ inline v16h load_a_frag(const h16* p, int kbA) {
  v8h lo = *(const v8h*)(p + kbA);
  v8h hi = *(const v8h*)(p + kbA + 16);
  return cat8(lo, hi);
}

// B fragment (32x16 f16, KxN) from NT-layout W[N,K]: lane holds column n,
// lanes 0-15 K={0..7,16..23}, lanes 16-31 K={8..15,24..31}.  p = &W[col*ldb + k0 + kbB]
static __device__ inline v16h load_b_frag(const h16* p) {
  v8h lo = *(const v8h*)p;
  v8h hi = *(const v8h*)(p + 16);
  return cat8(lo, hi);
}

// ---------------------------------------------------------------- convert
__global__ void cvt_f32_f16(const float* __restrict__ in, h16* __restrict__ out, int n) {
  int i = blockIdx.x * 256 + threadIdx.x;
  if (i < n) out[i] = (h16)in[i];
}

// ------------------------------------------------- positional softmax weights
// posw[b,h,t] = softmax_t( -dot(p[b,t,:], head_w[h,:]) )  (head_b cancels in softmax)
__global__ __launch_bounds__(256)
void pos_kernel(const float* __restrict__ pos, const float* __restrict__ w1,
                const float* __restrict__ b1, const float* __restrict__ w2,
                const float* __restrict__ b2, const float* __restrict__ hw,
                float* __restrict__ posw) {
  __shared__ float red[256];
  int t = threadIdx.x;
  int h = blockIdx.x & (HH - 1);
  int b = blockIdx.x >> 3;
  const float* pv = pos + ((size_t)(b * SS + t)) * PD;
  float hid[PD];
#pragma unroll
  for (int j = 0; j < PD; ++j) {
    float a = b1[j];
#pragma unroll
    for (int i = 0; i < PD; ++i) a += pv[i] * w1[j * PD + i];
    hid[j] = fmaxf(a, 0.f);
  }
  float ps = 0.f;
#pragma unroll
  for (int c = 0; c < P8; ++c) {
    float a = b2[c];
#pragma unroll
    for (int j = 0; j < PD; ++j) a += hid[j] * w2[c * PD + j];
    ps += a * hw[h * P8 + c];
  }
  float x = -ps;
  red[t] = x; __syncthreads();
  for (int o = 128; o > 0; o >>= 1) { if (t < o) red[t] = fmaxf(red[t], red[t + o]); __syncthreads(); }
  float mx = red[0]; __syncthreads();
  float e = __expf(x - mx);
  red[t] = e; __syncthreads();
  for (int o = 128; o > 0; o >>= 1) { if (t < o) red[t] += red[t + o]; __syncthreads(); }
  float sum = red[0];
  posw[(size_t)blockIdx.x * SS + t] = e / sum;
}

// ------------------------------------------------------------- GEMM (NT)
// C[M,N] = A[M,K] @ W[N,K]^T,  K == 256.
// Block: 8 waves x (32 rows x 64 cols) = 256x64 tile. B tile staged in LDS,
// A fragments double-buffered in registers so loads overlap WMMAs.
// mode 0: f16 row-major   mode 1: f16 V-transposed [B,H,HD,S]   mode 2: f32 + bias
__global__ __launch_bounds__(256)
void gemm_nt(const h16* __restrict__ A, const h16* __restrict__ W,
             h16* __restrict__ outH, float* __restrict__ outF,
             const float* __restrict__ bias, int M, int N, int K, int mode) {
  __shared__ __align__(16) h16 bs[64 * BSTR];  // 33,792 B
  int tid = threadIdx.x, lane = tid & 31, wave = tid >> 5;
  int l15 = lane & 15;
  int m0 = blockIdx.x * 256 + wave * 32;
  int n0 = blockIdx.y * 64;

  // Cooperative stage of W[n0..n0+63, 0..256) into LDS (2048 16-byte chunks)
  for (int idx = tid; idx < 64 * 32; idx += 256) {
    int col = idx >> 5, off = (idx & 31) * 8;
    *(v8h*)(bs + col * BSTR + off) = *(const v8h*)(W + (size_t)(n0 + col) * K + off);
  }
  __syncthreads();

  int kbA = (lane & 16) ? 8 : 0;
  int kbB = (lane & 16) ? 8 : 0;
  const h16* arow0 = A + (size_t)(m0 + l15) * K;
  const h16* arow1 = A + (size_t)(m0 + 16 + l15) * K;

  v8f acc[2][4] = {};
  v16h a0 = load_a_frag(arow0, kbA);
  v16h a1 = load_a_frag(arow1, kbA);

  for (int k0 = 0; k0 < K; k0 += 32) {
    int knext = (k0 + 32 < K) ? (k0 + 32) : 0;   // wrap: harmless extra load
    v16h na0 = load_a_frag(arow0 + knext, kbA);
    v16h na1 = load_a_frag(arow1 + knext, kbA);
#pragma unroll
    for (int t = 0; t < 4; ++t) {
      const h16* bp = bs + (size_t)(t * 16 + l15) * BSTR + k0 + kbB;
      v16h bf = load_b_frag(bp);
      acc[0][t] = __builtin_amdgcn_wmma_f32_16x16x32_f16(false, a0, false, bf,
                                                         (short)0, acc[0][t], false, false);
      acc[1][t] = __builtin_amdgcn_wmma_f32_16x16x32_f16(false, a1, false, bf,
                                                         (short)0, acc[1][t], false, false);
    }
    a0 = na0; a1 = na1;
  }

  int hi = (lane >> 4) & 1;
#pragma unroll
  for (int rg = 0; rg < 2; ++rg) {
#pragma unroll
    for (int t = 0; t < 4; ++t) {
#pragma unroll
      for (int j = 0; j < 8; ++j) {
        int r = m0 + rg * 16 + j + hi * 8;
        int c = n0 + t * 16 + l15;
        float v = acc[rg][t][j];
        if (mode == 2) {
          outF[(size_t)r * N + c] = v + bias[c];
        } else if (mode == 1) {
          int b = r >> 8, s = r & 255, h = c >> 5, d = c & 31;
          outH[((size_t)(b * HH + h) * HD + d) * SS + s] = (h16)v;
        } else {
          outH[(size_t)r * N + c] = (h16)v;
        }
      }
    }
  }
}

// ------------------------------------------------------------- fused attention
// grid = (S/64, H, B); 4 waves x 16 rows. q,k in [B,S,H,HD] (== [B*S, D] row-major),
// vT in [B,H,HD,S].  Writes head output f16 into oh[B*S, D].
__global__ __launch_bounds__(128)
void attn_kernel(const h16* __restrict__ q, const h16* __restrict__ k,
                 const h16* __restrict__ vT, const float* __restrict__ posw,
                 const float* __restrict__ gate, h16* __restrict__ oh) {
  __shared__ __align__(16) _Float16 slds[4 * 16 * BSTR];  // 33,792 B (padded rows)
  int tid = threadIdx.x, lane = tid & 31, wave = tid >> 5;
  int l15 = lane & 15, hi = (lane >> 4) & 1;
  int h = blockIdx.y, b = blockIdx.z;
  int m0 = blockIdx.x * 64 + wave * 16;
  int row = m0 + l15;
  float g = 1.f / (1.f + __expf(-gate[h]));
  const float scale = 0.17677669529663687f;  // 1/sqrt(32)
  int kbA = (lane & 16) ? 8 : 0;
  int kbB = (lane & 16) ? 8 : 0;

  // Q fragment: single K-chunk (HD == 32)
  const h16* qrow = q + (size_t)(b * SS + row) * DD + h * HD;
  v16h qa = load_a_frag(qrow, kbA);

  // Scores: one WMMA per 16x16 tile, 16 N-tiles across S=256; K frags prefetched.
  auto loadK = [&](int nt) -> v16h {
    const h16* kp = k + (size_t)(b * SS + nt * 16 + l15) * DD + h * HD + kbB;
    return load_b_frag(kp);
  };
  v8f acc[16];
  v8f zero = {};
  v16h kf = loadK(0);
#pragma unroll
  for (int nt = 0; nt < 16; ++nt) {
    v16h kn = loadK((nt + 1) & 15);  // wrap: redundant final load keeps code branch-free
    acc[nt] = __builtin_amdgcn_wmma_f32_16x16x32_f16(false, qa, false, kf,
                                                     (short)0, zero, false, false);
    kf = kn;
  }

  // Softmax over t (rows live at component j, halves of the wave)
#pragma unroll
  for (int nt = 0; nt < 16; ++nt)
#pragma unroll
    for (int i = 0; i < 8; ++i) acc[nt][i] *= scale;

  v8f pm = acc[0];
#pragma unroll
  for (int nt = 1; nt < 16; ++nt)
#pragma unroll
    for (int i = 0; i < 8; ++i) pm[i] = fmaxf(pm[i], acc[nt][i]);
#pragma unroll
  for (int off = 8; off >= 1; off >>= 1)
#pragma unroll
    for (int i = 0; i < 8; ++i) pm[i] = fmaxf(pm[i], __shfl_xor(pm[i], off, 32));

  v8f psum = {};
#pragma unroll
  for (int nt = 0; nt < 16; ++nt) {
#pragma unroll
    for (int i = 0; i < 8; ++i) acc[nt][i] = __expf(acc[nt][i] - pm[i]);
    psum += acc[nt];
  }
#pragma unroll
  for (int off = 8; off >= 1; off >>= 1)
#pragma unroll
    for (int i = 0; i < 8; ++i) psum[i] += __shfl_xor(psum[i], off, 32);

  v8f invp;
#pragma unroll
  for (int i = 0; i < 8; ++i) invp[i] = 1.f / psum[i];

  // Blend with positional weights + renormalize
  const float* pw = posw + (size_t)(b * HH + h) * SS;
  float omg = 1.f - g;
  v8f tsum = {};
#pragma unroll
  for (int nt = 0; nt < 16; ++nt) {
    float w = pw[nt * 16 + l15];
#pragma unroll
    for (int i = 0; i < 8; ++i) acc[nt][i] = omg * acc[nt][i] * invp[i] + g * w;
    tsum += acc[nt];
  }
#pragma unroll
  for (int off = 8; off >= 1; off >>= 1)
#pragma unroll
    for (int i = 0; i < 8; ++i) tsum[i] += __shfl_xor(tsum[i], off, 32);
  v8f rinv;
#pragma unroll
  for (int i = 0; i < 8; ++i) rinv[i] = 1.f / tsum[i];

  // C-fragment -> LDS (f16, padded stride) for A-fragment re-layout
  _Float16* my = slds + wave * 16 * BSTR;
#pragma unroll
  for (int nt = 0; nt < 16; ++nt)
#pragma unroll
    for (int j = 0; j < 8; ++j)
      my[(j + hi * 8) * BSTR + nt * 16 + l15] = (_Float16)(acc[nt][j] * rinv[j]);
  __syncthreads();

  // A . V  (K = 256 over t, N = 32 head dims), all frags prefetched one step ahead.
  const h16* vbase = vT + (size_t)(b * HH + h) * HD * SS;
  auto loadA = [&](int kt) -> v16h {
    return load_a_frag(my + l15 * BSTR + kt * 32, kbA);
  };
  auto loadV = [&](int kt, int n2) -> v16h {
    return load_b_frag(vbase + (size_t)(n2 * 16 + l15) * SS + kt * 32 + kbB);
  };
  v8f o[2] = {zero, zero};
  v16h af  = loadA(0);
  v16h bf0 = loadV(0, 0);
  v16h bf1 = loadV(0, 1);
#pragma unroll
  for (int kt = 0; kt < 16; ++kt) {
    int kn = (kt + 1) & 15;
    v16h naf  = loadA(kn);
    v16h nbf0 = loadV(kn, 0);
    v16h nbf1 = loadV(kn, 1);
    o[0] = __builtin_amdgcn_wmma_f32_16x16x32_f16(false, af, false, bf0,
                                                  (short)0, o[0], false, false);
    o[1] = __builtin_amdgcn_wmma_f32_16x16x32_f16(false, af, false, bf1,
                                                  (short)0, o[1], false, false);
    af = naf; bf0 = nbf0; bf1 = nbf1;
  }
#pragma unroll
  for (int n2 = 0; n2 < 2; ++n2)
#pragma unroll
    for (int j = 0; j < 8; ++j) {
      int s = m0 + j + hi * 8;
      int d = n2 * 16 + l15;
      oh[(size_t)(b * SS + s) * DD + h * HD + d] = (h16)o[n2][j];
    }
}

// ------------------------------------------------------------------ launch
extern "C" void kernel_launch(void* const* d_in, const int* in_sizes, int n_in,
                              void* d_out, int out_size, void* d_ws, size_t ws_size,
                              hipStream_t stream) {
  const float* x      = (const float*)d_in[0];
  const float* pos    = (const float*)d_in[1];
  const float* Wq     = (const float*)d_in[2];
  const float* Wk     = (const float*)d_in[3];
  const float* pw1    = (const float*)d_in[4];
  const float* pb1    = (const float*)d_in[5];
  const float* pw2    = (const float*)d_in[6];
  const float* pb2    = (const float*)d_in[7];
  const float* head_w = (const float*)d_in[8];
  const float* gate   = (const float*)d_in[10];
  const float* v_emb  = (const float*)d_in[11];
  const float* out_w  = (const float*)d_in[12];
  const float* out_b  = (const float*)d_in[13];

  char* ws = (char*)d_ws;
  const size_t XSZ = (size_t)MM * DD;        // 4,194,304 halfs
  const size_t WSZ = (size_t)DD * DD;        //    65,536 halfs
  h16* xh  = (h16*)ws;           ws += XSZ * 2;
  h16* Wqh = (h16*)ws;           ws += WSZ * 2;
  h16* Wkh = (h16*)ws;           ws += WSZ * 2;
  h16* Wvh = (h16*)ws;           ws += WSZ * 2;
  h16* Woh = (h16*)ws;           ws += WSZ * 2;
  h16* qh  = (h16*)ws;           ws += XSZ * 2;
  h16* kh  = (h16*)ws;           ws += XSZ * 2;
  h16* vTh = (h16*)ws;           ws += XSZ * 2;
  float* posw = (float*)ws;      ws += (size_t)BB * HH * SS * 4;
  h16* oh = xh;  // reuse: x no longer needed after QKV projections

  cvt_f32_f16<<<(XSZ + 255) / 256, 256, 0, stream>>>(x, xh, (int)XSZ);
  cvt_f32_f16<<<(WSZ + 255) / 256, 256, 0, stream>>>(Wq, Wqh, (int)WSZ);
  cvt_f32_f16<<<(WSZ + 255) / 256, 256, 0, stream>>>(Wk, Wkh, (int)WSZ);
  cvt_f32_f16<<<(WSZ + 255) / 256, 256, 0, stream>>>(v_emb, Wvh, (int)WSZ);
  cvt_f32_f16<<<(WSZ + 255) / 256, 256, 0, stream>>>(out_w, Woh, (int)WSZ);

  pos_kernel<<<BB * HH, 256, 0, stream>>>(pos, pw1, pb1, pw2, pb2, head_w, posw);

  dim3 gg(MM / 256, DD / 64);
  gemm_nt<<<gg, 256, 0, stream>>>(xh, Wqh, qh, nullptr, nullptr, MM, DD, DD, 0);
  gemm_nt<<<gg, 256, 0, stream>>>(xh, Wkh, kh, nullptr, nullptr, MM, DD, DD, 0);
  gemm_nt<<<gg, 256, 0, stream>>>(xh, Wvh, vTh, nullptr, nullptr, MM, DD, DD, 1);

  attn_kernel<<<dim3(SS / 64, HH, BB), 128, 0, stream>>>(qh, kh, vTh, posw, gate, oh);

  gemm_nt<<<gg, 256, 0, stream>>>(oh, Woh, nullptr, (float*)d_out, out_b, MM, DD, DD, 2);
}
